// S5Regressor_9354438771049
// MI455X (gfx1250) — compile-verified, matrix operand order checked by reading, not run on hardware
//
#include <hip/hip_runtime.h>

// ---------------------------------------------------------------------------
// S5 regressor forward for gfx1250 (MI455X).
// GEMMs on v_wmma_f32_16x16x32_bf16 (bf16 operands staged once by producers,
// fp32 accumulate, double-buffered LDS pipeline, 64x64 wave tiles);
// scan / layernorm / gelu / pooling stay fp32.
// ---------------------------------------------------------------------------

#define Hdim 2048
#define Pdim 2048
#define Bdim 8
#define Ldim 2048
#define Mrows (Bdim * Ldim)   // 16384 rows of activations

typedef __attribute__((ext_vector_type(16))) __bf16 v16bf;
typedef __attribute__((ext_vector_type(8)))  __bf16 v8bf;
typedef __attribute__((ext_vector_type(8)))  float  v8f;

__device__ __forceinline__ __bf16 f2bf(float x) {
  unsigned u = __builtin_bit_cast(unsigned, x);
  u += 0x7FFFu + ((u >> 16) & 1u);              // round-to-nearest-even
  unsigned short hb = (unsigned short)(u >> 16);
  return __builtin_bit_cast(__bf16, hb);
}

__device__ __forceinline__ float gelu_exact(float x) {
  return 0.5f * x * (1.0f + erff(x * 0.70710678118654752f));
}

// block of 256 threads (8 wave32); returns the full sum to every thread
__device__ __forceinline__ float block_sum256(float v, float* sbuf) {
#pragma unroll
  for (int o = 16; o > 0; o >>= 1) v += __shfl_xor(v, o, 32);
  int t = threadIdx.x;
  if ((t & 31) == 0) sbuf[t >> 5] = v;
  __syncthreads();
  float s = 0.f;
#pragma unroll
  for (int i = 0; i < 8; ++i) s += sbuf[i];
  __syncthreads();
  return s;
}

// ---------------------------------------------------------------------------
// K1: encoder (h = x * enc_w + enc_b) + LayerNorm -> fx (fp32 + bf16 copy).
// ---------------------------------------------------------------------------
__global__ __launch_bounds__(256) void enc_ln_kernel(
    const float* __restrict__ x, const float* __restrict__ enc_w,
    const float* __restrict__ enc_b, const float* __restrict__ g,
    const float* __restrict__ b, float* __restrict__ h_out,
    float* __restrict__ fx_out, __bf16* __restrict__ fx_bf) {
  __shared__ float sbuf[8];
  const int row = blockIdx.x;
  const float xv = x[row];
  float vals[8];
  float s = 0.f;
#pragma unroll
  for (int j = 0; j < 8; ++j) {
    int col = threadIdx.x + j * 256;
    float v = xv * enc_w[col] + enc_b[col];
    vals[j] = v;
    h_out[(size_t)row * Hdim + col] = v;
    s += v;
  }
  float mean = block_sum256(s, sbuf) * (1.f / Hdim);
  float q = 0.f;
#pragma unroll
  for (int j = 0; j < 8; ++j) { float d = vals[j] - mean; q += d * d; }
  float var = block_sum256(q, sbuf) * (1.f / Hdim);
  float rs = rsqrtf(var + 1e-5f);
#pragma unroll
  for (int j = 0; j < 8; ++j) {
    int col = threadIdx.x + j * 256;
    size_t o = (size_t)row * Hdim + col;
    float nv = (vals[j] - mean) * rs * g[col] + b[col];
    fx_out[o] = nv;
    fx_bf[o] = f2bf(nv);
  }
}

// ---------------------------------------------------------------------------
// K2: lambda_bar = exp(lam*step); s = (lambda_bar - 1)/lam    (per channel p)
// ---------------------------------------------------------------------------
__global__ void prep_lambda_kernel(
    const float* __restrict__ lam_re, const float* __restrict__ lam_im,
    const float* __restrict__ log_step, float* __restrict__ lbr,
    float* __restrict__ lbi, float* __restrict__ sre, float* __restrict__ sim) {
  int p = blockIdx.x * blockDim.x + threadIdx.x;
  if (p >= Pdim) return;
  float step = expf(log_step[p]);
  float lr = lam_re[p], li = lam_im[p];
  float er = expf(lr * step);
  float cbr = er * cosf(li * step);
  float cbi = er * sinf(li * step);
  lbr[p] = cbr;
  lbi[p] = cbi;
  float nr = cbr - 1.f, ni = cbi;
  float den = 1.f / (lr * lr + li * li);
  sre[p] = (nr * lr + ni * li) * den;
  sim[p] = (ni * lr - nr * li) * den;
}

// K3: b_bar[p,h] = s[p] * (B_re + i B_im)[p,h]  -> bf16 weights
__global__ void prep_bbar_kernel(
    const float* __restrict__ B_re, const float* __restrict__ B_im,
    const float* __restrict__ sre, const float* __restrict__ sim,
    __bf16* __restrict__ wr, __bf16* __restrict__ wi) {
  size_t stride = (size_t)gridDim.x * blockDim.x;
  size_t n = (size_t)Pdim * Hdim;
  for (size_t i = blockIdx.x * (size_t)blockDim.x + threadIdx.x; i < n; i += stride) {
    int p = (int)(i / Hdim);
    float br = B_re[i], bi = B_im[i];
    float a = sre[p], c = sim[p];
    wr[i] = f2bf(a * br - c * bi);
    wi[i] = f2bf(a * bi + c * br);
  }
}

// K4: fp32 -> bf16 weight conversion
__global__ void cvt_bf16_kernel(const float* __restrict__ src,
                                __bf16* __restrict__ dst, size_t n) {
  size_t stride = (size_t)gridDim.x * blockDim.x;
  for (size_t i = blockIdx.x * (size_t)blockDim.x + threadIdx.x; i < n; i += stride)
    dst[i] = f2bf(src[i]);
}

// ---------------------------------------------------------------------------
// Generic NT GEMM:  C[M,N] = alpha * A[M,K] * W[N,K]^T (+ addend) (+ C)
// A, W bf16; C fp32.  256 threads = 8 waves (2x4); block tile 128x256;
// wave tile 64x64 (4x4 WMMA accumulators, 16 WMMA per 32-K step, only 8
// fragment ds_loads per step).  Double-buffered LDS pipeline.
// M multiple of 128; N multiple of 256; K multiple of 32.
// ---------------------------------------------------------------------------
#define TM 128
#define TN 256
#define TK 32
#define LDT 48   // LDS row stride in halves (96B: 16B-aligned, bank-skewed)

__device__ __forceinline__ v16bf load_frag(const __bf16* s, int lane) {
  // ISA 16-bit A/B layout: lanes0-15 -> K0..7 / K16..23, lanes16-31 -> +8
  int r = lane & 15, hs = (lane >> 4) & 1;
  const __bf16* p = s + r * LDT + hs * 8;
  v8bf lo = *reinterpret_cast<const v8bf*>(p);
  v8bf hi = *reinterpret_cast<const v8bf*>(p + 16);
  return __builtin_shufflevector(lo, hi, 0, 1, 2, 3, 4, 5, 6, 7,
                                 8, 9, 10, 11, 12, 13, 14, 15);
}

__device__ __forceinline__ void compute_step(const __bf16* As, const __bf16* Bs,
                                             v8f acc[4][4], int wm, int wn,
                                             int lane) {
  v16bf af[4];
#pragma unroll
  for (int mt = 0; mt < 4; ++mt)
    af[mt] = load_frag(As + (wm * 64 + mt * 16) * LDT, lane);
#pragma unroll
  for (int nt = 0; nt < 4; ++nt) {
    v16bf bfrag = load_frag(Bs + (wn * 64 + nt * 16) * LDT, lane);
#pragma unroll
    for (int mt = 0; mt < 4; ++mt)
      acc[mt][nt] = __builtin_amdgcn_wmma_f32_16x16x32_bf16(
          false, af[mt], false, bfrag, (short)0, acc[mt][nt], false, false);
  }
}

__global__ __launch_bounds__(256) void gemm_bf16_nt(
    const __bf16* __restrict__ A, const __bf16* __restrict__ W,
    float* __restrict__ C, int M, int N, int K, float alpha,
    const float* __restrict__ addend, int accFlag) {
  __shared__ __align__(16) __bf16 As[2][TM * LDT];
  __shared__ __align__(16) __bf16 Bs[2][TN * LDT];

  const int t = threadIdx.x;
  const int lane = t & 31;
  const int wave = t >> 5;      // 0..7
  const int wm = wave >> 2;     // 0..1 (M direction, 64 rows each)
  const int wn = wave & 3;      // 0..3 (N direction, 64 cols each)
  const int m0 = blockIdx.y * TM;
  const int n0 = blockIdx.x * TN;

  // A staging: 2 threads per row, 16 halves (32B) each
  const int sra = t >> 1;
  const int sca = (t & 1) * 16;
  // B staging: 1 thread per row, 32 halves (64B) each
  const int srb = t;

  v8f acc[4][4];
#pragma unroll
  for (int mt = 0; mt < 4; ++mt)
#pragma unroll
    for (int nt = 0; nt < 4; ++nt)
#pragma unroll
      for (int i = 0; i < 8; ++i) acc[mt][nt][i] = 0.f;

  const __bf16* gA = A + (size_t)(m0 + sra) * K + sca;
  const __bf16* gW = W + (size_t)(n0 + srb) * K;
  __bf16* sa = &As[0][sra * LDT + sca];
  __bf16* sb = &Bs[0][srb * LDT];
  const int aStride = TM * LDT;
  const int bStride = TN * LDT;

  // prologue: stage k=0 into buffer 0
  {
    v8bf a0 = *reinterpret_cast<const v8bf*>(gA);
    v8bf a1 = *reinterpret_cast<const v8bf*>(gA + 8);
    v8bf w0 = *reinterpret_cast<const v8bf*>(gW);
    v8bf w1 = *reinterpret_cast<const v8bf*>(gW + 8);
    v8bf w2 = *reinterpret_cast<const v8bf*>(gW + 16);
    v8bf w3 = *reinterpret_cast<const v8bf*>(gW + 24);
    *reinterpret_cast<v8bf*>(sa) = a0;
    *reinterpret_cast<v8bf*>(sa + 8) = a1;
    *reinterpret_cast<v8bf*>(sb) = w0;
    *reinterpret_cast<v8bf*>(sb + 8) = w1;
    *reinterpret_cast<v8bf*>(sb + 16) = w2;
    *reinterpret_cast<v8bf*>(sb + 24) = w3;
  }
  __syncthreads();

  int cur = 0;
  for (int k0 = TK; k0 < K; k0 += TK) {
    // prefetch next K-slab into registers (overlaps with WMMA below)
    v8bf a0 = *reinterpret_cast<const v8bf*>(gA + k0);
    v8bf a1 = *reinterpret_cast<const v8bf*>(gA + k0 + 8);
    v8bf w0 = *reinterpret_cast<const v8bf*>(gW + k0);
    v8bf w1 = *reinterpret_cast<const v8bf*>(gW + k0 + 8);
    v8bf w2 = *reinterpret_cast<const v8bf*>(gW + k0 + 16);
    v8bf w3 = *reinterpret_cast<const v8bf*>(gW + k0 + 24);

    // compute on current buffer
    compute_step(&As[cur][0], &Bs[cur][0], acc, wm, wn, lane);

    // stage prefetched slab into the other buffer
    int nxt = cur ^ 1;
    __bf16* psa = sa + nxt * aStride;
    __bf16* psb = sb + nxt * bStride;
    *reinterpret_cast<v8bf*>(psa) = a0;
    *reinterpret_cast<v8bf*>(psa + 8) = a1;
    *reinterpret_cast<v8bf*>(psb) = w0;
    *reinterpret_cast<v8bf*>(psb + 8) = w1;
    *reinterpret_cast<v8bf*>(psb + 16) = w2;
    *reinterpret_cast<v8bf*>(psb + 24) = w3;
    __syncthreads();
    cur = nxt;
  }
  // final K-slab
  compute_step(&As[cur][0], &Bs[cur][0], acc, wm, wn, lane);

  // ---- epilogue ----
  const int col = lane & 15;
  const int rsel = (lane >> 4) & 1;   // VGPR i holds M=i (lanes<16) or M=8+i
#pragma unroll
  for (int mt = 0; mt < 4; ++mt) {
#pragma unroll
    for (int nt = 0; nt < 4; ++nt) {
      int cbase = n0 + wn * 64 + nt * 16 + col;
#pragma unroll
      for (int i = 0; i < 8; ++i) {
        int row = m0 + wm * 64 + mt * 16 + rsel * 8 + i;
        size_t o = (size_t)row * N + cbase;
        float v = alpha * acc[mt][nt][i];
        if (addend) v += addend[o];
        if (accFlag) v += C[o];
        C[o] = v;
      }
    }
  }
}

// ---------------------------------------------------------------------------
// K5: diagonal complex recurrence; reads fp32 bu, writes bf16 xs (GEMM food).
// One thread per (batch, p) sequence; coalesced across p each timestep.
// ---------------------------------------------------------------------------
__global__ __launch_bounds__(256) void scan_kernel(
    const float* __restrict__ lbr, const float* __restrict__ lbi,
    const float* __restrict__ bu_re, const float* __restrict__ bu_im,
    __bf16* __restrict__ xr_bf, __bf16* __restrict__ xi_bf) {
  int idx = blockIdx.x * blockDim.x + threadIdx.x;   // 0 .. B*P-1
  int b = idx / Pdim;
  int p = idx - b * Pdim;
  float ar = lbr[p], ai = lbi[p];
  float xr = 0.f, xi = 0.f;
  size_t base = (size_t)b * Ldim * Pdim + p;
#pragma unroll 4
  for (int t = 0; t < Ldim; ++t) {
    size_t o = base + (size_t)t * Pdim;
    float br = bu_re[o], bi = bu_im[o];
    float nr = fmaf(ar, xr, fmaf(-ai, xi, br));
    float ni = fmaf(ar, xi, fmaf(ai, xr, bi));
    xr = nr; xi = ni;
    xr_bf[o] = f2bf(xr);
    xi_bf[o] = f2bf(xi);
  }
}

// ---------------------------------------------------------------------------
// K6: z = gelu(acc + D*fx) + fx  ; fz = LN(z, ff_g, ff_b)  (fp32 + bf16 copy)
// (acc already holds 2*Re(xs C^T) via GEMM alphas +2 / -2.)
// ---------------------------------------------------------------------------
__global__ __launch_bounds__(256) void post_s5_kernel(
    const float* __restrict__ acc, const float* __restrict__ fx,
    const float* __restrict__ D, const float* __restrict__ g,
    const float* __restrict__ b, float* __restrict__ fz,
    __bf16* __restrict__ fz_bf) {
  __shared__ float sbuf[8];
  const int row = blockIdx.x;
  float vals[8];
  float s = 0.f;
#pragma unroll
  for (int j = 0; j < 8; ++j) {
    int col = threadIdx.x + j * 256;
    size_t o = (size_t)row * Hdim + col;
    float fxv = fx[o];
    float y = acc[o] + D[col] * fxv;
    float z = gelu_exact(y) + fxv;
    vals[j] = z;
    s += z;
  }
  float mean = block_sum256(s, sbuf) * (1.f / Hdim);
  float q = 0.f;
#pragma unroll
  for (int j = 0; j < 8; ++j) { float d = vals[j] - mean; q += d * d; }
  float var = block_sum256(q, sbuf) * (1.f / Hdim);
  float rs = rsqrtf(var + 1e-5f);
#pragma unroll
  for (int j = 0; j < 8; ++j) {
    int col = threadIdx.x + j * 256;
    size_t o = (size_t)row * Hdim + col;
    float nv = (vals[j] - mean) * rs * g[col] + b[col];
    fz[o] = nv;
    fz_bf[o] = f2bf(nv);
  }
}

// K7: GEGLU  ge = a * gelu(g) -> bf16 only (feeds the decoder GEMM)
__global__ void geglu_kernel(const float* __restrict__ ff,
                             __bf16* __restrict__ ge_bf) {
  size_t stride = (size_t)gridDim.x * blockDim.x;
  size_t n = (size_t)Mrows * Hdim;
  for (size_t i = blockIdx.x * (size_t)blockDim.x + threadIdx.x; i < n; i += stride) {
    size_t row = i / Hdim;
    int col = (int)(i - row * Hdim);
    float a = ff[row * 2 * Hdim + col];
    float gg = ff[row * 2 * Hdim + Hdim + col];
    ge_bf[i] = f2bf(a * gelu_exact(gg));
  }
}

// K8: r = h + z2; LN(out_g, out_b); srow = dot(norm, head_w).  Block per row.
__global__ __launch_bounds__(256) void final_row_kernel(
    const float* __restrict__ h, const float* __restrict__ z2,
    const float* __restrict__ g, const float* __restrict__ b,
    const float* __restrict__ head_w, float* __restrict__ srow) {
  __shared__ float sbuf[8];
  const int row = blockIdx.x;
  float vals[8];
  float s = 0.f;
#pragma unroll
  for (int j = 0; j < 8; ++j) {
    int col = threadIdx.x + j * 256;
    size_t o = (size_t)row * Hdim + col;
    float v = h[o] + z2[o];
    vals[j] = v;
    s += v;
  }
  float mean = block_sum256(s, sbuf) * (1.f / Hdim);
  float q = 0.f;
#pragma unroll
  for (int j = 0; j < 8; ++j) { float d = vals[j] - mean; q += d * d; }
  float var = block_sum256(q, sbuf) * (1.f / Hdim);
  float rs = rsqrtf(var + 1e-5f);
  float dot = 0.f;
#pragma unroll
  for (int j = 0; j < 8; ++j) {
    int col = threadIdx.x + j * 256;
    float nv = (vals[j] - mean) * rs * g[col] + b[col];
    dot += nv * head_w[col];
  }
  float total = block_sum256(dot, sbuf);
  if (threadIdx.x == 0) srow[row] = total;
}

// K9: out[b] = mean_t srow[b,t] + head_b
__global__ __launch_bounds__(256) void pool_head_kernel(
    const float* __restrict__ srow, const float* __restrict__ head_b,
    float* __restrict__ out) {
  __shared__ float sbuf[8];
  int b = blockIdx.x;
  float s = 0.f;
  for (int t = threadIdx.x; t < Ldim; t += 256) s += srow[(size_t)b * Ldim + t];
  float total = block_sum256(s, sbuf);
  if (threadIdx.x == 0) out[b] = total * (1.f / Ldim) + head_b[0];
}

// ---------------------------------------------------------------------------
extern "C" void kernel_launch(void* const* d_in, const int* in_sizes, int n_in,
                              void* d_out, int out_size, void* d_ws,
                              size_t ws_size, hipStream_t stream) {
  (void)in_sizes; (void)n_in; (void)out_size; (void)ws_size;
  const float* x       = (const float*)d_in[0];
  const float* enc_w   = (const float*)d_in[1];
  const float* enc_b   = (const float*)d_in[2];
  const float* attn_g  = (const float*)d_in[3];
  const float* attn_b  = (const float*)d_in[4];
  const float* lam_re  = (const float*)d_in[5];
  const float* lam_im  = (const float*)d_in[6];
  const float* log_step= (const float*)d_in[7];
  const float* B_re    = (const float*)d_in[8];
  const float* B_im    = (const float*)d_in[9];
  const float* C_re    = (const float*)d_in[10];
  const float* C_im    = (const float*)d_in[11];
  const float* Dv      = (const float*)d_in[12];
  const float* ff_g    = (const float*)d_in[13];
  const float* ff_b    = (const float*)d_in[14];
  const float* ff_enc  = (const float*)d_in[15];
  const float* ff_dec  = (const float*)d_in[16];
  const float* out_g   = (const float*)d_in[17];
  const float* out_b   = (const float*)d_in[18];
  const float* head_w  = (const float*)d_in[19];
  const float* head_b  = (const float*)d_in[20];
  float* out = (float*)d_out;

  const size_t MH = (size_t)Mrows * Hdim;
  const size_t PH = (size_t)Pdim * Hdim;
  char* ws = (char*)d_ws;
  size_t off = 0;
  auto alloc = [&](size_t bytes) -> char* {
    char* p = ws + off;
    off = (off + bytes + 255) & ~(size_t)255;
    return p;
  };
  float*  h_buf  = (float*)alloc(MH * 4);
  float*  fx_buf = (float*)alloc(MH * 4);
  __bf16* fx_bf  = (__bf16*)alloc(MH * 2);
  float*  bu_re  = (float*)alloc(MH * 4);   // -> reused as z2
  float*  bu_im  = (float*)alloc(MH * 4);   // -> reused as acc
  __bf16* xsr_bf = (__bf16*)alloc(MH * 2);
  __bf16* xsi_bf = (__bf16*)alloc(MH * 2);
  float*  fz_buf = (float*)alloc(MH * 4);
  __bf16* fz_bf  = (__bf16*)alloc(MH * 2);
  float*  ff_buf = (float*)alloc(2 * MH * 4);
  __bf16* ge_bf  = (__bf16*)alloc(MH * 2);
  __bf16* wbbr   = (__bf16*)alloc(PH * 2);
  __bf16* wbbi   = (__bf16*)alloc(PH * 2);
  __bf16* wcr    = (__bf16*)alloc(PH * 2);
  __bf16* wci    = (__bf16*)alloc(PH * 2);
  __bf16* wfe    = (__bf16*)alloc(2 * PH * 2);
  __bf16* wfd    = (__bf16*)alloc(PH * 2);
  float*  lbr    = (float*)alloc(Pdim * 4);
  float*  lbi    = (float*)alloc(Pdim * 4);
  float*  sre    = (float*)alloc(Pdim * 4);
  float*  sim    = (float*)alloc(Pdim * 4);
  float*  srow   = (float*)alloc((size_t)Mrows * 4);
  float*  acc    = bu_im;    // reuse: bu_im consumed by scan before y-GEMMs
  float*  z2     = bu_re;    // reuse: bu_re consumed by scan before z2-GEMM

  // K1: encoder + LN
  enc_ln_kernel<<<Mrows, 256, 0, stream>>>(x, enc_w, enc_b, attn_g, attn_b,
                                           h_buf, fx_buf, fx_bf);
  // K2-K4: weight prep (fp32 math, bf16 storage)
  prep_lambda_kernel<<<(Pdim + 255) / 256, 256, 0, stream>>>(
      lam_re, lam_im, log_step, lbr, lbi, sre, sim);
  prep_bbar_kernel<<<2048, 256, 0, stream>>>(B_re, B_im, sre, sim, wbbr, wbbi);
  cvt_bf16_kernel<<<2048, 256, 0, stream>>>(C_re, wcr, PH);
  cvt_bf16_kernel<<<2048, 256, 0, stream>>>(C_im, wci, PH);
  cvt_bf16_kernel<<<2048, 256, 0, stream>>>(ff_enc, wfe, 2 * PH);
  cvt_bf16_kernel<<<2048, 256, 0, stream>>>(ff_dec, wfd, PH);

  // GEMM 1/2: bu = fx @ b_bar^T  (re, im)
  gemm_bf16_nt<<<dim3(Pdim / TN, Mrows / TM), 256, 0, stream>>>(
      fx_bf, wbbr, bu_re, Mrows, Pdim, Hdim, 1.f, nullptr, 0);
  gemm_bf16_nt<<<dim3(Pdim / TN, Mrows / TM), 256, 0, stream>>>(
      fx_bf, wbbi, bu_im, Mrows, Pdim, Hdim, 1.f, nullptr, 0);

  // K5: complex diagonal scan (fp32 state, bf16 outputs for the GEMMs)
  scan_kernel<<<(Bdim * Pdim) / 256, 256, 0, stream>>>(lbr, lbi, bu_re, bu_im,
                                                       xsr_bf, xsi_bf);

  // GEMM 3/4: acc = 2*(xs_re C_re^T) - 2*(xs_im C_im^T)
  gemm_bf16_nt<<<dim3(Hdim / TN, Mrows / TM), 256, 0, stream>>>(
      xsr_bf, wcr, acc, Mrows, Hdim, Pdim, 2.f, nullptr, 0);
  gemm_bf16_nt<<<dim3(Hdim / TN, Mrows / TM), 256, 0, stream>>>(
      xsi_bf, wci, acc, Mrows, Hdim, Pdim, -2.f, nullptr, 1);

  // K6: z = gelu(acc + D*fx) + fx ; fz = LN(z)
  post_s5_kernel<<<Mrows, 256, 0, stream>>>(acc, fx_buf, Dv, ff_g, ff_b,
                                            fz_buf, fz_bf);

  // GEMM 5: ff = fz @ ff_enc^T  [M, 2H]
  gemm_bf16_nt<<<dim3(2 * Hdim / TN, Mrows / TM), 256, 0, stream>>>(
      fz_bf, wfe, ff_buf, Mrows, 2 * Hdim, Hdim, 1.f, nullptr, 0);

  // K7: GEGLU -> bf16
  geglu_kernel<<<2048, 256, 0, stream>>>(ff_buf, ge_bf);

  // GEMM 6: z2 = ge @ ff_dec^T + fz
  gemm_bf16_nt<<<dim3(Hdim / TN, Mrows / TM), 256, 0, stream>>>(
      ge_bf, wfd, z2, Mrows, Hdim, Hdim, 1.f, fz_buf, 0);

  // K8/K9: LN(h + z2) -> dot(head_w) per row, then mean over L + head_b
  final_row_kernel<<<Mrows, 256, 0, stream>>>(h_buf, z2, out_g, out_b, head_w,
                                              srow);
  pool_head_kernel<<<Bdim, 256, 0, stream>>>(srow, head_b, out);
}
